// SwinBlock_67637144978020
// MI455X (gfx1250) — matrix operand order repaired and checked by reading, compiled, not run on hardware
//
#include <hip/hip_runtime.h>
#include <hip/hip_bf16.h>

// ---------------- CDNA5 WMMA helpers (wave32, 16x16x32 f16 -> f32) ----------------
typedef __attribute__((ext_vector_type(16))) _Float16 v16h;
typedef __attribute__((ext_vector_type(8)))  float    f32x8;

// A fragment: 16x32 (MxK). lane l: row m = l&15; elements e<8 -> K=k0+e+8*hi,
// e>=8 -> K=k0+16+(e-8)+8*hi  => two contiguous 16B chunks.
__device__ __forceinline__ v16h load_a_frag(const _Float16* base, int stride,
                                            int m0, int k0, int lane) {
    int m = lane & 15, hi = lane >> 4;
    const _Float16* p = base + (size_t)(m0 + m) * stride + k0 + 8 * hi;
    union { uint4 q[2]; v16h v; } u;
    u.q[0] = *(const uint4*)(p);
    u.q[1] = *(const uint4*)(p + 16);
    return u.v;
}

// B fragment: 32x16 (KxN). lane l: col n = l&15; elements e<8 -> K=k0+e+8*hi,
// e>=8 -> K=k0+16+(e-8)+8*hi  => two contiguous 16B chunks (source is row-major [N][K]).
__device__ __forceinline__ v16h load_b_frag(const _Float16* base, int stride,
                                            int n0, int k0, int lane) {
    int n = lane & 15, hi = lane >> 4;
    const _Float16* p = base + (size_t)(n0 + n) * stride + k0 + 8 * hi;
    union { uint4 q[2]; v16h v; } u;
    u.q[0] = *(const uint4*)(p);
    u.q[1] = *(const uint4*)(p + 16);
    return u.v;
}

__device__ __forceinline__ f32x8 wmma16(v16h a, v16h b, f32x8 c) {
    return __builtin_amdgcn_wmma_f32_16x16x32_f16(false, a, false, b, (short)0, c,
                                                  false, false);
}

// ---------------- constants ----------------
#define BDIM   2
#define CDIM   96
#define HDIM   512
#define WDIM   512
#define HWN    (HDIM * WDIM)
#define NWIN   8192          // 2 * 64 * 64 windows
#define NTOK   64            // tokens per window
#define HEADS  3
#define HD     32
#define SCALE  0.17677669529663687f   // 32^-0.5

// ---------------- K0: weight f32 -> f16 ----------------
__global__ __launch_bounds__(256) void k_cvt(const float* __restrict__ s,
                                             _Float16* __restrict__ d, int n) {
    int i = blockIdx.x * 256 + threadIdx.x;
    if (i < n) d[i] = (_Float16)s[i];
}

// ---------------- K1: LN1 + band shift + window partition -> WIN f16 [8192][64][96]
__global__ __launch_bounds__(256) void k_ln1_win(const float* __restrict__ x,
                                                 const float* __restrict__ g,
                                                 const float* __restrict__ b,
                                                 _Float16* __restrict__ win) {
    int widx = blockIdx.x;
    int bb = widx >> 12, wh = (widx >> 6) & 63, ww = widx & 63;
    int sgn4 = (wh & 1) ? -4 : 4;
    __shared__ __align__(16) float xs[64][97];
    __shared__ float mean_s[64], rstd_s[64];
    for (int e = threadIdx.x; e < 64 * 96; e += 256) {
        int c = e >> 6, n = e & 63;
        int h = wh * 8 + (n >> 3);
        int wv = ww * 8 + (n & 7);
        int ws_ = (wv - sgn4 + 512) & 511;
        xs[n][c] = x[(((size_t)bb * 96 + c) * 512 + h) * 512 + ws_];
    }
    __syncthreads();
    if (threadIdx.x < 64) {
        int n = threadIdx.x;
        float s = 0.f, ss = 0.f;
        for (int c = 0; c < 96; ++c) { float v = xs[n][c]; s += v; ss += v * v; }
        float m = s * (1.f / 96.f);
        float var = ss * (1.f / 96.f) - m * m;
        mean_s[n] = m;
        rstd_s[n] = rsqrtf(var + 1e-5f);
    }
    __syncthreads();
    _Float16* wp = win + (size_t)widx * (64 * 96);
    for (int e = threadIdx.x; e < 64 * 96; e += 256) {
        int n = e / 96, c = e % 96;
        float v = (xs[n][c] - mean_s[n]) * rstd_s[n] * g[c] + b[c];
        wp[e] = (_Float16)v;
    }
}

// ---------------- K2: QKV GEMM per window ----------------
__global__ __launch_bounds__(256) void k_qkv(const _Float16* __restrict__ win,
                                             const _Float16* __restrict__ wq,
                                             const float* __restrict__ qb,
                                             _Float16* __restrict__ Q,
                                             _Float16* __restrict__ K,
                                             _Float16* __restrict__ VT) {
    int widx = blockIdx.x;
    int lane = threadIdx.x & 31, wave = threadIdx.x >> 5;
    const _Float16* A = win + (size_t)widx * (64 * 96);
    for (int t = wave; t < 72; t += 8) {          // 4 m-tiles x 18 oc-tiles
        int m0 = (t / 18) * 16, oc0 = (t % 18) * 16;
        f32x8 acc = {};
#pragma unroll
        for (int kc = 0; kc < 3; ++kc) {
            v16h a = load_a_frag(A, 96, m0, kc * 32, lane);
            v16h bf = load_b_frag(wq, 96, oc0, kc * 32, lane);
            acc = wmma16(a, bf, acc);
        }
        // segment / head / d-base are wave-uniform (oc0 multiple of 16; the q/k/v
        // boundary 96 and head boundary 32 are multiples of 16)
        int seg  = oc0 / 96;            // uniform: 0=q 1=k 2=v
        int rem0 = oc0 - seg * 96;      // uniform
        int head = rem0 >> 5;           // uniform
        int d    = (rem0 & 31) + (lane & 15);
        int hi   = lane >> 4;
        int oc   = oc0 + (lane & 15);
        float bias = qb[oc];
        size_t hb = ((size_t)widx * 3 + head);
        if (seg == 0) {
#pragma unroll
            for (int r = 0; r < 8; ++r) {
                int n = m0 + r + 8 * hi;
                Q[(hb * 64 + n) * 32 + d] = (_Float16)((acc[r] + bias) * SCALE);
            }
        } else if (seg == 1) {
#pragma unroll
            for (int r = 0; r < 8; ++r) {
                int n = m0 + r + 8 * hi;
                K[(hb * 64 + n) * 32 + d] = (_Float16)(acc[r] + bias);
            }
        } else {
#pragma unroll
            for (int r = 0; r < 8; ++r) {
                int n = m0 + r + 8 * hi;
                VT[(hb * 32 + d) * 64 + n] = (_Float16)(acc[r] + bias);
            }
        }
    }
}

// ---------------- K3: attention per (window, head): 1 wave per task ----------------
__global__ __launch_bounds__(128) void k_attn(const _Float16* __restrict__ Q,
                                              const _Float16* __restrict__ K,
                                              const _Float16* __restrict__ VT,
                                              const float* __restrict__ lapA,
                                              const float* __restrict__ lapB,
                                              _Float16* __restrict__ AO) {
    __shared__ __align__(16) float    S[4][64 * 65];
    __shared__ __align__(16) _Float16 P[4][64 * 72];
    int lane = threadIdx.x & 31, wave = threadIdx.x >> 5;
    int task = blockIdx.x * 4 + wave;
    int widx = task / 3, head = task % 3;
    const _Float16* Qp = Q  + ((size_t)widx * 3 + head) * 2048;
    const _Float16* Kp = K  + ((size_t)widx * 3 + head) * 2048;
    const _Float16* Vp = VT + ((size_t)widx * 3 + head) * 2048;
    float*    Sl = S[wave];
    _Float16* Pl = P[wave];

    // S = Q * K^T  (64x64, K-dim = 32 -> single WMMA per tile)
#pragma unroll
    for (int mt = 0; mt < 4; ++mt) {
        v16h a = load_a_frag(Qp, 32, mt * 16, 0, lane);
#pragma unroll
        for (int jt = 0; jt < 4; ++jt) {
            v16h bf = load_b_frag(Kp, 32, jt * 16, 0, lane);
            f32x8 acc = {};
            acc = wmma16(a, bf, acc);
            int col = jt * 16 + (lane & 15), hi = lane >> 4;
#pragma unroll
            for (int r = 0; r < 8; ++r)
                Sl[(mt * 16 + r + 8 * hi) * 65 + col] = acc[r];
        }
    }
    __syncthreads();

    // softmax rows (each lane owns rows lane and lane+32) with Laplacian bias
    float A2 = (*lapA) * (*lapA);
    float iB2 = 1.0f / ((*lapB) * (*lapB));
#pragma unroll
    for (int rr = 0; rr < 2; ++rr) {
        int row = lane + rr * 32;
        int r1 = row >> 3, c1 = row & 7;
        float mx = -1e30f;
        for (int c = 0; c < 64; ++c) {
            int rel = abs(r1 - (c >> 3)) + abs(c1 - (c & 7));
            float v = Sl[row * 65 + c] + A2 * __expf(-0.5f * (float)rel * iB2);
            Sl[row * 65 + c] = v;
            mx = fmaxf(mx, v);
        }
        float sum = 0.f;
        for (int c = 0; c < 64; ++c) {
            float e = __expf(Sl[row * 65 + c] - mx);
            Sl[row * 65 + c] = e;
            sum += e;
        }
        float inv = 1.0f / sum;
        for (int c = 0; c < 64; ++c)
            Pl[row * 72 + c] = (_Float16)(Sl[row * 65 + c] * inv);
    }
    __syncthreads();

    // O = P * V  (64x32, K-dim = 64 -> two WMMA steps)
    _Float16* Op = AO + (size_t)widx * (64 * 96) + head * 32;
#pragma unroll
    for (int mt = 0; mt < 4; ++mt) {
#pragma unroll
        for (int dt = 0; dt < 2; ++dt) {
            f32x8 acc = {};
#pragma unroll
            for (int kc = 0; kc < 2; ++kc) {
                v16h a  = load_a_frag(Pl, 72, mt * 16, kc * 32, lane);
                v16h bf = load_b_frag(Vp, 64, dt * 16, kc * 32, lane);
                acc = wmma16(a, bf, acc);
            }
            int d = dt * 16 + (lane & 15), hi = lane >> 4;
#pragma unroll
            for (int r = 0; r < 8; ++r) {
                int n = mt * 16 + r + 8 * hi;
                Op[(size_t)n * 96 + d] = (_Float16)acc[r];
            }
        }
    }
}

// ---------------- K4: proj + window reverse + inverse shift + residual -> XT f32
__global__ __launch_bounds__(256) void k_proj(const _Float16* __restrict__ AO,
                                              const _Float16* __restrict__ wp,
                                              const float* __restrict__ pb,
                                              const float* __restrict__ x,
                                              float* __restrict__ xt) {
    int widx = blockIdx.x;
    int bb = widx >> 12, wh = (widx >> 6) & 63, ww = widx & 63;
    int sgn4 = (wh & 1) ? -4 : 4;
    int lane = threadIdx.x & 31, wave = threadIdx.x >> 5;
    const _Float16* A = AO + (size_t)widx * (64 * 96);
    for (int t = wave; t < 24; t += 8) {          // 4 x 6 tiles
        int m0 = (t / 6) * 16, oc0 = (t % 6) * 16;
        int oc = oc0 + (lane & 15), hi = lane >> 4;

        // token -> (h, w_final) mapping for this lane's 8 result rows
        int hrow[8], wfin[8];
#pragma unroll
        for (int r = 0; r < 8; ++r) {
            int n = m0 + r + 8 * hi;
            hrow[r] = wh * 8 + (n >> 3);
            int wv = ww * 8 + (n & 7);
            wfin[r] = (wv - sgn4 + 512) & 511;
        }
        // prefetch the strided residual lines so they overlap the WMMA chain
        const float* xc = x + ((size_t)bb * 96 + oc) * HWN;
#pragma unroll
        for (int r = 0; r < 8; ++r)
            __builtin_prefetch(&xc[(size_t)hrow[r] * 512 + wfin[r]], 0, 3);

        f32x8 acc = {};
#pragma unroll
        for (int kc = 0; kc < 3; ++kc) {
            v16h a  = load_a_frag(A, 96, m0, kc * 32, lane);
            v16h bf = load_b_frag(wp, 96, oc0, kc * 32, lane);
            acc = wmma16(a, bf, acc);
        }
        float bias = pb[oc];
#pragma unroll
        for (int r = 0; r < 8; ++r) {
            size_t tok = ((size_t)bb * 512 + hrow[r]) * 512 + wfin[r];
            float v = acc[r] + bias + xc[(size_t)hrow[r] * 512 + wfin[r]];
            xt[tok * 96 + oc] = v;
        }
    }
}

// ---------------- K5: fused LN2 + MLP1 + GELU + MLP2 + residual + NCHW store ------
__global__ __launch_bounds__(256) void k_mlp(const float* __restrict__ xt,
                                             const float* __restrict__ g2,
                                             const float* __restrict__ b2,
                                             const _Float16* __restrict__ w1,
                                             const float* __restrict__ mb1,
                                             const _Float16* __restrict__ w2,
                                             const float* __restrict__ mb2,
                                             float* __restrict__ out) {
    __shared__ __align__(16) _Float16 xn[64 * 104];
    __shared__ __align__(16) _Float16 h1[64 * 392];
    __shared__ __align__(16) float    ob[96 * 66];
    int lane = threadIdx.x & 31, wave = threadIdx.x >> 5;
    size_t t0 = (size_t)blockIdx.x * 64;

    // LN2: one wave handles 8 tokens; lane covers channels lane, lane+32, lane+64
    for (int i = 0; i < 8; ++i) {
        int tl = wave * 8 + i;
        const float* row = xt + (t0 + tl) * 96;
        float v0 = row[lane], v1 = row[lane + 32], v2 = row[lane + 64];
        float s = v0 + v1 + v2;
        for (int o = 16; o; o >>= 1) s += __shfl_xor(s, o, 32);
        float q = v0 * v0 + v1 * v1 + v2 * v2;
        for (int o = 16; o; o >>= 1) q += __shfl_xor(q, o, 32);
        float m = s * (1.f / 96.f);
        float var = q * (1.f / 96.f) - m * m;
        float rs = rsqrtf(var + 1e-5f);
        xn[tl * 104 + lane]      = (_Float16)((v0 - m) * rs * g2[lane]      + b2[lane]);
        xn[tl * 104 + lane + 32] = (_Float16)((v1 - m) * rs * g2[lane + 32] + b2[lane + 32]);
        xn[tl * 104 + lane + 64] = (_Float16)((v2 - m) * rs * g2[lane + 64] + b2[lane + 64]);
    }
    __syncthreads();

    // GEMM1 (64x96 @ 96x384) + GELU -> h1 (LDS-resident)
    for (int t = wave; t < 96; t += 8) {          // 4 x 24 tiles
        int m0 = (t / 24) * 16, oc0 = (t % 24) * 16;
        f32x8 acc = {};
#pragma unroll
        for (int kc = 0; kc < 3; ++kc) {
            v16h a  = load_a_frag(xn, 104, m0, kc * 32, lane);
            v16h bf = load_b_frag(w1, 96, oc0, kc * 32, lane);
            acc = wmma16(a, bf, acc);
        }
        int oc = oc0 + (lane & 15), hi = lane >> 4;
        float bias = mb1[oc];
#pragma unroll
        for (int r = 0; r < 8; ++r) {
            float v = acc[r] + bias;
            v = 0.5f * v * (1.0f + erff(v * 0.70710678118f));
            h1[(m0 + r + 8 * hi) * 392 + oc] = (_Float16)v;
        }
    }
    __syncthreads();

    // GEMM2 (64x384 @ 384x96) + bias + residual -> ob (transposed in LDS)
    for (int t = wave; t < 24; t += 8) {          // 4 x 6 tiles
        int m0 = (t / 6) * 16, oc0 = (t % 6) * 16;
        f32x8 acc = {};
#pragma unroll
        for (int kc = 0; kc < 12; ++kc) {
            v16h a  = load_a_frag(h1, 392, m0, kc * 32, lane);
            v16h bf = load_b_frag(w2, 384, oc0, kc * 32, lane);
            acc = wmma16(a, bf, acc);
        }
        int oc = oc0 + (lane & 15), hi = lane >> 4;
        float bias = mb2[oc];
#pragma unroll
        for (int r = 0; r < 8; ++r) {
            int n = m0 + r + 8 * hi;
            ob[oc * 66 + n] = acc[r] + bias + xt[(t0 + n) * 96 + oc];
        }
    }
    __syncthreads();

    // Coalesced NCHW store (64 consecutive w per block)
    size_t b_ = t0 >> 18;
    size_t rem = t0 & (HWN - 1);
    int h = (int)(rem >> 9), w0 = (int)(rem & 511);
    for (int e = threadIdx.x; e < 96 * 64; e += 256) {
        int c = e >> 6, wq = e & 63;
        out[(((size_t)b_ * 96 + c) * 512 + h) * 512 + w0 + wq] = ob[c * 66 + wq];
    }
}

// ---------------- host launcher ----------------
extern "C" void kernel_launch(void* const* d_in, const int* in_sizes, int n_in,
                              void* d_out, int out_size, void* d_ws, size_t ws_size,
                              hipStream_t stream) {
    const float* x      = (const float*)d_in[0];
    const float* ln1_g  = (const float*)d_in[1];
    const float* ln1_b  = (const float*)d_in[2];
    const float* qkv_w  = (const float*)d_in[3];
    const float* qkv_b  = (const float*)d_in[4];
    const float* lap_A  = (const float*)d_in[5];
    const float* lap_B  = (const float*)d_in[6];
    const float* proj_w = (const float*)d_in[7];
    const float* proj_b = (const float*)d_in[8];
    const float* ln2_g  = (const float*)d_in[9];
    const float* ln2_b  = (const float*)d_in[10];
    const float* mlp_w1 = (const float*)d_in[11];
    const float* mlp_b1 = (const float*)d_in[12];
    const float* mlp_w2 = (const float*)d_in[13];
    const float* mlp_b2 = (const float*)d_in[14];
    float* out = (float*)d_out;

    char* ws = (char*)d_ws;
    // f16 weights
    _Float16* WQ = (_Float16*)(ws + 0);                     // 27648 elems
    _Float16* WP = (_Float16*)(ws + 55296);                 //  9216
    _Float16* W1 = (_Float16*)(ws + 73728);                 // 36864
    _Float16* W2 = (_Float16*)(ws + 147456);                // 36864
    // big buffers
    const size_t SZ_WIN = (size_t)NWIN * 64 * 96 * sizeof(_Float16);     // 96 MB
    const size_t SZ_QKV = (size_t)NWIN * 3 * 64 * 32 * sizeof(_Float16); // 96 MB each
    _Float16* WIN = (_Float16*)(ws + 262144);
    _Float16* Qb  = (_Float16*)(ws + 262144 + SZ_WIN);
    _Float16* Kb  = (_Float16*)(ws + 262144 + SZ_WIN + SZ_QKV);
    _Float16* VTb = (_Float16*)(ws + 262144 + SZ_WIN + 2 * SZ_QKV);
    float*    XT  = (float*)   (ws + 262144 + SZ_WIN + 3 * SZ_QKV);      // 192 MB
    _Float16* AO  = WIN;   // reuse WIN region for attention output

    // K0: weight conversion (tiny, L2-resident afterwards)
    k_cvt<<<(27648 + 255) / 256, 256, 0, stream>>>(qkv_w,  WQ, 27648);
    k_cvt<<<(9216  + 255) / 256, 256, 0, stream>>>(proj_w, WP, 9216);
    k_cvt<<<(36864 + 255) / 256, 256, 0, stream>>>(mlp_w1, W1, 36864);
    k_cvt<<<(36864 + 255) / 256, 256, 0, stream>>>(mlp_w2, W2, 36864);

    // K1: LN1 + shift + window partition
    k_ln1_win<<<NWIN, 256, 0, stream>>>(x, ln1_g, ln1_b, WIN);
    // K2: QKV GEMM
    k_qkv<<<NWIN, 256, 0, stream>>>(WIN, WQ, qkv_b, Qb, Kb, VTb);
    // K3: windowed attention (24576 wave-tasks, 4 per block)
    k_attn<<<(NWIN * HEADS) / 4, 128, 0, stream>>>(Qb, Kb, VTb, lap_A, lap_B, AO);
    // K4: projection + reverse shift + residual
    k_proj<<<NWIN, 256, 0, stream>>>(AO, WP, proj_b, x, XT);
    // K5: fused LN2 + MLP + residual + transpose store
    k_mlp<<<(BDIM * HWN) / 64, 256, 0, stream>>>(XT, ln2_g, ln2_b, W1, mlp_b1,
                                                 W2, mlp_b2, out);
}
